// DecoderGQALayer_1443109011687
// MI455X (gfx1250) — matrix operand order repaired and checked by reading, not compiled, not run on hardware
//
#include <hip/hip_runtime.h>

// ---------------- problem constants ----------------
#define BB  2
#define SS  1024
#define DDIM 1024
#define GG  8
#define HH  2
#define HDIM 64
#define EE  8
#define FFD 4096
#define TOK (BB * SS)          // 2048 rows

// ---------------- CDNA5 async-to-LDS availability ----------------
#if defined(__has_builtin)
#  if __has_builtin(__builtin_amdgcn_global_load_async_to_lds_b128) && \
      __has_builtin(__builtin_amdgcn_s_wait_asynccnt)
#    define USE_ASYNC_LDS 1
#  endif
#endif
#ifndef USE_ASYNC_LDS
#  define USE_ASYNC_LDS 0
#endif

// Builtin takes pointers to 16B int vectors (per hipcc diagnostic).
typedef int v4i __attribute__((vector_size(16)));
typedef __attribute__((address_space(1))) v4i* as1_v4i_p;
typedef __attribute__((address_space(3))) v4i* as3_v4i_p;

// ---------------- WMMA types ----------------
typedef __attribute__((ext_vector_type(16))) __bf16 v16bf;
typedef __attribute__((ext_vector_type(8)))  float  v8f;

union BF16Frag { v16bf v; unsigned int u[8]; unsigned short s[16]; };
union F8Frag   { v8f   v; float f[8]; };

// ---------------- bf16 helpers (manual RNE) ----------------
__device__ __forceinline__ unsigned short f2bf(float f) {
    union { float f; unsigned u; } c; c.f = f;
    unsigned b = c.u;
    b += 0x7FFFu + ((b >> 16) & 1u);
    return (unsigned short)(b >> 16);
}
__device__ __forceinline__ float bf2f(unsigned short h) {
    union { unsigned u; float f; } c; c.u = ((unsigned)h) << 16;
    return c.f;
}

__device__ __forceinline__ float redmax16(float v) {
    #pragma unroll
    for (int m = 1; m < 16; m <<= 1) v = fmaxf(v, __shfl_xor(v, m, 16));
    return v;
}
__device__ __forceinline__ float redsum16(float v) {
    #pragma unroll
    for (int m = 1; m < 16; m <<= 1) v += __shfl_xor(v, m, 16);
    return v;
}

// =====================================================================
// RMSNorm (f32 in) -> bf16 out.  One block (256 thr) per row of D=1024.
// =====================================================================
__global__ __launch_bounds__(256)
void rmsnorm_bf16_k(const float* __restrict__ in, const float* __restrict__ w,
                    unsigned short* __restrict__ out) {
    const int row = blockIdx.x;
    const int tid = threadIdx.x;
    const float* xr = in + (size_t)row * DDIM;
    float s = 0.f;
    #pragma unroll
    for (int i = 0; i < 4; ++i) { float v = xr[tid + i * 256]; s += v * v; }
    #pragma unroll
    for (int m = 1; m < 32; m <<= 1) s += __shfl_xor(s, m, 32);
    __shared__ float ps[8];
    if ((tid & 31) == 0) ps[tid >> 5] = s;
    __syncthreads();
    float tot = 0.f;
    #pragma unroll
    for (int i = 0; i < 8; ++i) tot += ps[i];
    const float r = rsqrtf(tot / (float)DDIM + 1e-6f);
    #pragma unroll
    for (int i = 0; i < 4; ++i) {
        int c = tid + i * 256;
        out[(size_t)row * DDIM + c] = f2bf(w[c] * xr[c] * r);
    }
}

// =====================================================================
// WMMA GEMM:  C[M,N](f32) (+)= A[M,K](bf16 row major) * B[K,N](f32->bf16)
// 256 thr = 8 waves; tile BM=128 x BN; K-step 32; A tile double-buffered
// through the CDNA5 async-to-LDS path (fallback: sync vector copies).
// rowcnt != null  : tiles with m0 >= *rowcnt exit (sparse experts).
// EPI 0: C[m,n] = acc                 (skips rows >= *rowcnt)
// EPI 1: C[scat_idx[m], n] += scat_w[m] * acc   (token scatter-accumulate)
// =====================================================================
template <int BN, int EPI>
__global__ __launch_bounds__(256)
void gemm_wmma_k(const unsigned short* __restrict__ A, int lda, int K,
                 const float* __restrict__ B0, long long bZstride, int ldb,
                 float* __restrict__ C0, int ldc, int cColStride,
                 const int* __restrict__ rowcnt,
                 const int* __restrict__ scat_idx,
                 const float* __restrict__ scat_w) {
    constexpr int BM = 128, BK = 32, BKP = BK + 8, NT = BN / 16;
    __shared__ unsigned short sA[2][BM][BKP];    // padded: 80B row stride
    __shared__ unsigned short sB[BN][BKP];       // transposed [n][k]

    const int tid  = threadIdx.x;
    const int lane = tid & 31;
    const int wv   = tid >> 5;
    const int lo   = lane & 15;
    const int hi   = lane >> 4;
    const int m0   = blockIdx.y * BM;
    const int n0   = blockIdx.x * BN;

    int cntv = 0x7fffffff;
    if (rowcnt) { cntv = *rowcnt; if (m0 >= cntv) return; }

    const float* __restrict__ Bsrc = B0 + (long long)blockIdx.z * bZstride;
    float* __restrict__ Cp = C0 + (long long)blockIdx.z * cColStride;

    auto stageA = [&](int kk, int buf) {
        for (int i = tid; i < BM * 4; i += 256) {        // 2 b128 per thread
            int r = i >> 2, c = i & 3;
            const unsigned short* g = A + (long long)(m0 + r) * lda + kk + c * 8;
            unsigned short* l = &sA[buf][r][c * 8];
#if USE_ASYNC_LDS
            __builtin_amdgcn_global_load_async_to_lds_b128(
                (as1_v4i_p)(void*)const_cast<unsigned short*>(g),
                (as3_v4i_p)(void*)l, 0, 0);
#else
            *(uint4*)l = *(const uint4*)g;
#endif
        }
    };

    F8Frag acc[NT];
    #pragma unroll
    for (int t = 0; t < NT; ++t)
        #pragma unroll
        for (int j = 0; j < 8; ++j) acc[t].f[j] = 0.f;

    stageA(0, 0);                                        // pipeline prologue
    int ib = 0;
    for (int kk = 0; kk < K; kk += BK, ib ^= 1) {
        // ---- stage B tile (f32 -> bf16, transposed); coalesced reads ----
        constexpr int NSHIFT = (BN == 128) ? 7 : 6;
        for (int i = tid; i < BN * BK; i += 256) {
            int n = i & (BN - 1), k = i >> NSHIFT;
            sB[n][k] = f2bf(Bsrc[(long long)(kk + k) * ldb + n0 + n]);
        }
        const bool more = (kk + BK < K);
        if (more) {
            stageA(kk + BK, ib ^ 1);                     // prefetch next A tile
            __builtin_prefetch(&Bsrc[(long long)(kk + BK) * ldb + n0 + (tid & (BN - 1))], 0, 1);
        }
#if USE_ASYNC_LDS
        if (more) __builtin_amdgcn_s_wait_asynccnt(2);   // current tile landed
        else      __builtin_amdgcn_s_wait_asynccnt(0);
#endif
        __syncthreads();

        // ---- A fragment: ISA 16-bit A 16x32 layout ----
        const int mrow = wv * 16 + lo;
        BF16Frag af;
        #pragma unroll
        for (int p = 0; p < 8; ++p) {
            int k = 2 * p + 8 * hi + ((p >= 4) ? 8 : 0);
            af.u[p] = *(const unsigned int*)&sA[ib][mrow][k];
        }
        // ---- N tiles ----
        #pragma unroll
        for (int t = 0; t < NT; ++t) {
            BF16Frag bfr;
            const int n = t * 16 + lo;
            #pragma unroll
            for (int p = 0; p < 8; ++p) {
                int k = 2 * p + 8 * hi + ((p >= 4) ? 8 : 0); // B 32x16 layout
                bfr.u[p] = *(const unsigned int*)&sB[n][k];
            }
            acc[t].v = __builtin_amdgcn_wmma_f32_16x16x32_bf16(
                false, af.v, false, bfr.v, (short)0, acc[t].v, false, false);
        }
        __syncthreads();
    }

    // ---- epilogue (C/D layout: lane lo = N, vgpr j -> M = j + 8*hi) ----
    #pragma unroll
    for (int t = 0; t < NT; ++t) {
        const int n = n0 + t * 16 + lo;
        #pragma unroll
        for (int j = 0; j < 8; ++j) {
            const int m = m0 + wv * 16 + j + 8 * hi;
            float v = acc[t].f[j];
            if (EPI == 0) {
                if (m < cntv) Cp[(long long)m * ldc + n] = v;
            } else {
                if (m < cntv) {
                    const int tok = scat_idx[m];
                    Cp[(long long)tok * ldc + n] += scat_w[m] * v;
                }
            }
        }
    }
}

// =====================================================================
// RoPE on Q ([b,s,g,h,hd] f32 -> [b,g,h,s,hd] bf16)
// =====================================================================
__global__ __launch_bounds__(256)
void rope_q_k(const float* __restrict__ qf, unsigned short* __restrict__ qb) {
    long long idx = (long long)blockIdx.x * 256 + threadIdx.x;   // B*S*G*H*32
    int i = idx & 31; long long t = idx >> 5;
    int h = t % HH; t /= HH;
    int g = t % GG; t /= GG;
    int s = t % SS; int b = t / SS;
    float inv = __expf(-((float)(2 * i) / 64.f) * logf(10000.f));
    float ang = (float)s * inv, c, sn;
    __sincosf(ang, &sn, &c);
    long long rbase = ((((long long)b * SS + s) * GG + g) * HH + h) * 64;
    float x1 = qf[rbase + i], x2 = qf[rbase + 32 + i];
    long long wbase = ((((long long)b * GG + g) * HH + h) * SS + s) * 64;
    qb[wbase + i]      = f2bf(x1 * c - x2 * sn);
    qb[wbase + 32 + i] = f2bf(x1 * sn + x2 * c);
}

// RoPE on K ([b,s,g,hd] f32 -> [b,g,s,hd] bf16)
__global__ __launch_bounds__(256)
void rope_k_k(const float* __restrict__ kf, unsigned short* __restrict__ kb) {
    long long idx = (long long)blockIdx.x * 256 + threadIdx.x;   // B*S*G*32
    int i = idx & 31; long long t = idx >> 5;
    int g = t % GG; t /= GG;
    int s = t % SS; int b = t / SS;
    float inv = __expf(-((float)(2 * i) / 64.f) * logf(10000.f));
    float ang = (float)s * inv, c, sn;
    __sincosf(ang, &sn, &c);
    long long rbase = (((long long)b * SS + s) * (GG * HDIM)) + g * 64;
    float x1 = kf[rbase + i], x2 = kf[rbase + 32 + i];
    long long wbase = (((long long)b * GG + g) * SS + s) * 64;
    kb[wbase + i]      = f2bf(x1 * c - x2 * sn);
    kb[wbase + 32 + i] = f2bf(x1 * sn + x2 * c);
}

// V reorder/convert ([b,s,g,hd] f32 -> [b,g,s,hd] bf16)
__global__ __launch_bounds__(256)
void conv_v_k(const float* __restrict__ vf, unsigned short* __restrict__ vb) {
    long long idx = (long long)blockIdx.x * 256 + threadIdx.x;   // B*S*G*64
    int i = idx & 63; long long t = idx >> 6;
    int g = t % GG; t /= GG;
    int s = t % SS; int b = t / SS;
    long long rbase = (((long long)b * SS + s) * (GG * HDIM)) + g * 64;
    long long wbase = (((long long)b * GG + g) * SS + s) * 64;
    vb[wbase + i] = f2bf(vf[rbase + i]);
}

// =====================================================================
// Flash attention, WMMA bf16, causal, one wave per 16-query tile.
// Writes x1 = x + attn_out.
// =====================================================================
__global__ __launch_bounds__(32)
void flash_attn_k(const unsigned short* __restrict__ qb,   // [B,G,H,S,HD]
                  const unsigned short* __restrict__ kb,   // [B,G,S,HD]
                  const unsigned short* __restrict__ vb,   // [B,G,S,HD]
                  const float* __restrict__ x, float* __restrict__ x1) {
    const int qt = blockIdx.x;
    const int gh = blockIdx.y;
    const int b  = blockIdx.z;
    const int g  = gh / HH, h = gh % HH;
    const int lane = threadIdx.x;
    const int lo = lane & 15, hi = lane >> 4;
    const int q0 = qt * 16;

    __shared__ unsigned short pbuf[16][32];

    const unsigned short* Q  = qb + ((((long long)b * GG + g) * HH + h) * SS + q0) * 64;
    const unsigned short* Kp = kb + (((long long)b * GG + g)) * SS * 64;
    const unsigned short* Vp = vb + (((long long)b * GG + g)) * SS * 64;

    BF16Frag qa[2];
    #pragma unroll
    for (int half = 0; half < 2; ++half)
        #pragma unroll
        for (int p = 0; p < 8; ++p) {
            int kd = half * 32 + 2 * p + 8 * hi + ((p >= 4) ? 8 : 0);
            qa[half].u[p] = *(const unsigned int*)(Q + (long long)lo * 64 + kd);
        }

    float mrun[8], lrun[8];
    F8Frag o[4];
    #pragma unroll
    for (int j = 0; j < 8; ++j) { mrun[j] = -1e30f; lrun[j] = 0.f; }
    #pragma unroll
    for (int c = 0; c < 4; ++c)
        #pragma unroll
        for (int j = 0; j < 8; ++j) o[c].f[j] = 0.f;

    const int kend = q0 + 16;
    for (int k0 = 0; k0 < kend; k0 += 32) {
        F8Frag sc[2];
        #pragma unroll
        for (int sub = 0; sub < 2; ++sub) {
            #pragma unroll
            for (int j = 0; j < 8; ++j) sc[sub].f[j] = 0.f;
            const int key = k0 + sub * 16 + lo;
            BF16Frag kf0, kf1;
            #pragma unroll
            for (int p = 0; p < 8; ++p) {
                int kd = 2 * p + 8 * hi + ((p >= 4) ? 8 : 0);
                kf0.u[p] = *(const unsigned int*)(Kp + (long long)key * 64 + kd);
                kf1.u[p] = *(const unsigned int*)(Kp + (long long)key * 64 + 32 + kd);
            }
            sc[sub].v = __builtin_amdgcn_wmma_f32_16x16x32_bf16(
                false, qa[0].v, false, kf0.v, (short)0, sc[sub].v, false, false);
            sc[sub].v = __builtin_amdgcn_wmma_f32_16x16x32_bf16(
                false, qa[1].v, false, kf1.v, (short)0, sc[sub].v, false, false);
        }
        #pragma unroll
        for (int j = 0; j < 8; ++j) {
            const int row = q0 + j + 8 * hi;
            #pragma unroll
            for (int sub = 0; sub < 2; ++sub) {
                const int key = k0 + sub * 16 + lo;
                float v = sc[sub].f[j] * 0.125f;   // 1/sqrt(64)
                if (key > row) v = -1e30f;
                sc[sub].f[j] = v;
            }
            float mn = redmax16(fmaxf(sc[0].f[j], sc[1].f[j]));
            mn = fmaxf(mrun[j], mn);
            float alpha = __expf(mrun[j] - mn);
            lrun[j] *= alpha;
            #pragma unroll
            for (int c = 0; c < 4; ++c) o[c].f[j] *= alpha;
            float p0 = __expf(sc[0].f[j] - mn);
            float p1 = __expf(sc[1].f[j] - mn);
            lrun[j] += redsum16(p0 + p1);
            mrun[j] = mn;
            sc[0].f[j] = p0; sc[1].f[j] = p1;
        }
        #pragma unroll
        for (int j = 0; j < 8; ++j) {
            pbuf[j + 8 * hi][lo]      = f2bf(sc[0].f[j]);
            pbuf[j + 8 * hi][16 + lo] = f2bf(sc[1].f[j]);
        }
        __syncthreads();
        BF16Frag pa;
        #pragma unroll
        for (int p = 0; p < 8; ++p) {
            int k = 2 * p + 8 * hi + ((p >= 4) ? 8 : 0);
            pa.u[p] = *(const unsigned int*)&pbuf[lo][k];
        }
        #pragma unroll
        for (int c = 0; c < 4; ++c) {
            BF16Frag vfr;
            #pragma unroll
            for (int p = 0; p < 8; ++p) {
                const int kA = k0 + 2 * p + 8 * hi + ((p >= 4) ? 8 : 0);
                unsigned short a0 = Vp[(long long)kA * 64 + c * 16 + lo];
                unsigned short a1 = Vp[(long long)(kA + 1) * 64 + c * 16 + lo];
                vfr.u[p] = (unsigned)a0 | ((unsigned)a1 << 16);
            }
            o[c].v = __builtin_amdgcn_wmma_f32_16x16x32_bf16(
                false, pa.v, false, vfr.v, (short)0, o[c].v, false, false);
        }
        __syncthreads();
    }

    #pragma unroll
    for (int c = 0; c < 4; ++c)
        #pragma unroll
        for (int j = 0; j < 8; ++j) {
            const int row = q0 + j + 8 * hi;
            const int col = (g * HH + h) * 64 + c * 16 + lo;
            const long long idx = ((long long)b * SS + row) * DDIM + col;
            x1[idx] = x[idx] + o[c].f[j] / lrun[j];
        }
}

// =====================================================================
// Router: softmax(E=8), top-2 renormalized, token compaction per expert.
// =====================================================================
__global__ __launch_bounds__(32)
void router_k(const unsigned short* __restrict__ h2, const float* __restrict__ rw,
              int* __restrict__ cnt, int* __restrict__ idxl,
              float* __restrict__ wgt, float* __restrict__ probs) {
    const int t = blockIdx.x;
    const int lane = threadIdx.x;
    float lg[EE];
    #pragma unroll
    for (int e = 0; e < EE; ++e) lg[e] = 0.f;
    for (int d = lane; d < DDIM; d += 32) {
        float hv = bf2f(h2[(size_t)t * DDIM + d]);
        #pragma unroll
        for (int e = 0; e < EE; ++e) lg[e] += hv * rw[d * EE + e];
    }
    #pragma unroll
    for (int e = 0; e < EE; ++e)
        #pragma unroll
        for (int m = 1; m < 32; m <<= 1) lg[e] += __shfl_xor(lg[e], m, 32);

    float mx = lg[0];
    #pragma unroll
    for (int e = 1; e < EE; ++e) mx = fmaxf(mx, lg[e]);
    float p[EE], den = 0.f;
    #pragma unroll
    for (int e = 0; e < EE; ++e) { p[e] = __expf(lg[e] - mx); den += p[e]; }
    #pragma unroll
    for (int e = 0; e < EE; ++e) p[e] /= den;

    int i1 = 0;
    #pragma unroll
    for (int e = 1; e < EE; ++e) if (p[e] > p[i1]) i1 = e;
    int i2 = (i1 == 0) ? 1 : 0;
    #pragma unroll
    for (int e = 0; e < EE; ++e) if (e != i1 && p[e] > p[i2]) i2 = e;
    const float wsum = p[i1] + p[i2];

    if (lane == 0) {
        #pragma unroll
        for (int e = 0; e < EE; ++e) probs[t * EE + e] = p[e];
        int s1 = atomicAdd(&cnt[i1], 1);
        idxl[i1 * TOK + s1] = t; wgt[i1 * TOK + s1] = p[i1] / wsum;
        int s2 = atomicAdd(&cnt[i2], 1);
        idxl[i2 * TOK + s2] = t; wgt[i2 * TOK + s2] = p[i2] / wsum;
    }
}

// Gather selected tokens into per-expert contiguous bf16 activation rows.
__global__ __launch_bounds__(256)
void gather_k(const unsigned short* __restrict__ h2b, const int* __restrict__ cnt,
              const int* __restrict__ idxl, unsigned short* __restrict__ ha) {
    const int e = blockIdx.x / TOK, slot = blockIdx.x % TOK;
    if (slot >= cnt[e]) return;
    const int tok = idxl[e * TOK + slot];
    const uint2* src = (const uint2*)(h2b + (size_t)tok * DDIM);
    uint2* dst = (uint2*)(ha + ((size_t)e * TOK + slot) * DDIM);
    dst[threadIdx.x] = src[threadIdx.x];       // 256 thr x 8B = 2048B row
}

// Deterministic aux loss: fixed-order tree reduction of router probs.
__global__ __launch_bounds__(256)
void aux_final_k(const int* __restrict__ cnt, const float* __restrict__ probs,
                 float* __restrict__ out) {
    __shared__ float red[256];
    const int tid = threadIdx.x;
    float ps[EE];
    #pragma unroll
    for (int e = 0; e < EE; ++e) ps[e] = 0.f;
    for (int t = tid; t < TOK; t += 256)
        #pragma unroll
        for (int e = 0; e < EE; ++e) ps[e] += probs[t * EE + e];
    float aux = 0.f;
    for (int e = 0; e < EE; ++e) {
        red[tid] = ps[e];
        __syncthreads();
        for (int st = 128; st > 0; st >>= 1) {
            if (tid < st) red[tid] += red[tid + st];
            __syncthreads();
        }
        if (tid == 0) {
            float f = (float)cnt[e] / (float)(TOK * 2);
            aux += f * (red[0] / (float)TOK);
        }
        __syncthreads();
    }
    if (tid == 0) out[(size_t)TOK * DDIM] = (float)EE * aux;
}

// SwiGLU: actb = bf16( silu(g) * u )
__global__ __launch_bounds__(256)
void silu_mul_k(const float* __restrict__ g, const float* __restrict__ u,
                unsigned short* __restrict__ out, long long n) {
    long long i = (long long)blockIdx.x * 256 + threadIdx.x;
    if (i >= n) return;
    float gv = g[i];
    float a = gv / (1.f + __expf(-gv));
    out[i] = f2bf(a * u[i]);
}

// out = x1 + moe_acc
__global__ __launch_bounds__(256)
void final_add_k(const float* __restrict__ x1, const float* __restrict__ macc,
                 float* __restrict__ out, long long n) {
    long long i = (long long)blockIdx.x * 256 + threadIdx.x;
    if (i >= n) return;
    out[i] = x1[i] + macc[i];
}

// =====================================================================
// Host orchestration
// =====================================================================
extern "C" void kernel_launch(void* const* d_in, const int* in_sizes, int n_in,
                              void* d_out, int out_size, void* d_ws, size_t ws_size,
                              hipStream_t stream) {
    (void)in_sizes; (void)n_in; (void)out_size; (void)ws_size;
    const float* x        = (const float*)d_in[0];
    const float* norm1_w  = (const float*)d_in[1];
    const float* Wq       = (const float*)d_in[2];
    const float* Wk       = (const float*)d_in[3];
    const float* Wv       = (const float*)d_in[4];
    const float* norm2_w  = (const float*)d_in[5];
    const float* router_w = (const float*)d_in[6];
    const float* w_gate   = (const float*)d_in[7];
    const float* w_up     = (const float*)d_in[8];
    const float* w_down   = (const float*)d_in[9];
    float* out = (float*)d_out;

    // ---- workspace carve-up (256B aligned regions) ----
    char* wsp = (char*)d_ws;
    auto carve = [&](size_t bytes) -> char* {
        char* p = wsp;
        wsp += (bytes + 255) & ~(size_t)255;
        return p;
    };
    unsigned short* hx   = (unsigned short*)carve((size_t)TOK * DDIM * 2);
    float*          qf   = (float*)carve((size_t)TOK * DDIM * 4);
    float*          kf   = (float*)carve((size_t)TOK * (GG * HDIM) * 4);
    float*          vf   = (float*)carve((size_t)TOK * (GG * HDIM) * 4);
    unsigned short* qb   = (unsigned short*)carve((size_t)TOK * DDIM * 2);
    unsigned short* kb   = (unsigned short*)carve((size_t)TOK * (GG * HDIM) * 2);
    unsigned short* vb   = (unsigned short*)carve((size_t)TOK * (GG * HDIM) * 2);
    float*          x1   = (float*)carve((size_t)TOK * DDIM * 4);
    unsigned short* h2b  = (unsigned short*)carve((size_t)TOK * DDIM * 2);
    int*            cnt  = (int*)carve(EE * 4);
    int*            idxl = (int*)carve((size_t)EE * TOK * 4);
    float*          wgt  = (float*)carve((size_t)EE * TOK * 4);
    float*         probs = (float*)carve((size_t)TOK * EE * 4);
    unsigned short* ha   = (unsigned short*)carve((size_t)EE * TOK * DDIM * 2);
    float*          actg = (float*)carve((size_t)TOK * FFD * 4);
    float*          actu = (float*)carve((size_t)TOK * FFD * 4);
    unsigned short* actb = (unsigned short*)carve((size_t)TOK * FFD * 2);
    float*          macc = (float*)carve((size_t)TOK * DDIM * 4);

    // ---- 1. norm1 -> bf16 ----
    rmsnorm_bf16_k<<<TOK, 256, 0, stream>>>(x, norm1_w, hx);

    // ---- 2. grouped QKV projections (WMMA bf16, grid.z = group) ----
    gemm_wmma_k<128, 0><<<dim3(1, TOK / 128, GG), 256, 0, stream>>>(
        hx, DDIM, DDIM, Wq, (long long)DDIM * 128, 128, qf, DDIM, 128,
        nullptr, nullptr, nullptr);
    gemm_wmma_k<64, 0><<<dim3(1, TOK / 128, GG), 256, 0, stream>>>(
        hx, DDIM, DDIM, Wk, (long long)DDIM * 64, 64, kf, GG * HDIM, 64,
        nullptr, nullptr, nullptr);
    gemm_wmma_k<64, 0><<<dim3(1, TOK / 128, GG), 256, 0, stream>>>(
        hx, DDIM, DDIM, Wv, (long long)DDIM * 64, 64, vf, GG * HDIM, 64,
        nullptr, nullptr, nullptr);

    // ---- 3. RoPE + layout to [b,g,h,s,hd] bf16 ----
    rope_q_k<<<(TOK * GG * HH * 32) / 256, 256, 0, stream>>>(qf, qb);
    rope_k_k<<<(TOK * GG * 32) / 256, 256, 0, stream>>>(kf, kb);
    conv_v_k<<<(TOK * GG * 64) / 256, 256, 0, stream>>>(vf, vb);

    // ---- 4. flash attention, residual add -> x1 ----
    flash_attn_k<<<dim3(SS / 16, GG * HH, BB), 32, 0, stream>>>(qb, kb, vb, x, x1);

    // ---- 5. norm2 -> bf16 ----
    rmsnorm_bf16_k<<<TOK, 256, 0, stream>>>(x1, norm2_w, h2b);

    // ---- 6. router, token compaction, deterministic aux loss ----
    (void)hipMemsetAsync(cnt, 0, EE * 4, stream);
    (void)hipMemsetAsync(macc, 0, (size_t)TOK * DDIM * 4, stream);
    router_k<<<TOK, 32, 0, stream>>>(h2b, router_w, cnt, idxl, wgt, probs);
    gather_k<<<EE * TOK, 256, 0, stream>>>(h2b, cnt, idxl, ha);
    aux_final_k<<<1, 256, 0, stream>>>(cnt, probs, out);

    // ---- 7. sparse experts: compacted rows, 128-row tiles early-exit ----
    const long long nact = (long long)TOK * FFD;
    for (int e = 0; e < EE; ++e) {
        const unsigned short* hae = ha + (size_t)e * TOK * DDIM;
        gemm_wmma_k<128, 0><<<dim3(FFD / 128, TOK / 128, 1), 256, 0, stream>>>(
            hae, DDIM, DDIM, w_gate + (long long)e * DDIM * FFD, 0, FFD,
            actg, FFD, 0, cnt + e, nullptr, nullptr);
        gemm_wmma_k<128, 0><<<dim3(FFD / 128, TOK / 128, 1), 256, 0, stream>>>(
            hae, DDIM, DDIM, w_up + (long long)e * DDIM * FFD, 0, FFD,
            actu, FFD, 0, cnt + e, nullptr, nullptr);
        silu_mul_k<<<(nact + 255) / 256, 256, 0, stream>>>(actg, actu, actb, nact);
        gemm_wmma_k<128, 1><<<dim3(DDIM / 128, TOK / 128, 1), 256, 0, stream>>>(
            actb, FFD, FFD, w_down + (long long)e * FFD * DDIM, 0, DDIM,
            macc, DDIM, 0, cnt + e, idxl + (size_t)e * TOK, wgt + (size_t)e * TOK);
    }

    // ---- 8. out = x1 + moe ----
    const long long nout = (long long)TOK * DDIM;
    final_add_k<<<(nout + 255) / 256, 256, 0, stream>>>(x1, macc, out, nout);
}
